// GlobalPointer_25409026523722
// MI455X (gfx1250) — matrix operand order repaired and checked by reading, not compiled, not run on hardware
//
#include <hip/hip_runtime.h>

// Problem constants (from reference)
#define HID   1024
#define NCOL  2048        // ENT_TYPE * INNER_DIM * 2
#define BB    8
#define LL    512
#define HH    16          // ENT_TYPE
#define DD    64          // INNER_DIM
#define NEGV  1e12f

typedef __attribute__((ext_vector_type(16))) _Float16 v16h;
typedef __attribute__((ext_vector_type(8)))  _Float16 v8h;
typedef __attribute__((ext_vector_type(8)))  float    v8f;
typedef unsigned int  u32x4 __attribute__((ext_vector_type(4)));
typedef int           i32x8 __attribute__((ext_vector_type(8)));
typedef int           i32x4 __attribute__((ext_vector_type(4)));

union Frag16 { v16h v; v8h h[2]; _Float16 e[16]; };

// --- Tensor Data Mover availability (6-arg clang-23 form, per diagnostic) ---
#if defined(__has_builtin)
#  if __has_builtin(__builtin_amdgcn_tensor_load_to_lds) && \
      __has_builtin(__builtin_amdgcn_s_wait_tensorcnt)
#    define USE_TDM 1
#  endif
#endif
#ifndef USE_TDM
#  define USE_TDM 0
#endif

#if USE_TDM
// Build a 2D f32 tile descriptor per CDNA5 ISA and issue
// TENSOR_LOAD_TO_LDS. All inputs are wave-uniform -> descriptor lives in SGPRs.
//   tile_d0 x tile_d1 f32 elements, row stride = stride_elems (f32 units),
//   packed row-major into LDS at byte offset lds_off.
__device__ __forceinline__ void tdm_load_tile_f32(
    unsigned lds_off, const float* gptr,
    unsigned tile_d0, unsigned tile_d1,
    unsigned tens_d0, unsigned tens_d1, unsigned stride_elems)
{
    unsigned long long ga = (unsigned long long)(uintptr_t)gptr;
    u32x4 g0;
    g0[0] = 1u;                                        // count=1, no gather
    g0[1] = lds_off;                                   // lds_addr [63:32]
    g0[2] = (unsigned)(ga & 0xFFFFFFFFu);              // global_addr lo
    g0[3] = (unsigned)((ga >> 32) & 0x1FFFFFFu)        // global_addr hi (57b)
          | (2u << 30);                                // type=2 ("image")
    i32x8 g1;
    g1[0] = (int)(2u << 16);                           // data_size=2 -> 4 bytes
    g1[1] = (int)((tens_d0 & 0xFFFFu) << 16);          // tensor_dim0[15:0]
    g1[2] = (int)(((tens_d0 >> 16) & 0xFFFFu)          // tensor_dim0[31:16]
          | ((tens_d1 & 0xFFFFu) << 16));              // tensor_dim1[15:0]
    g1[3] = (int)(((tens_d1 >> 16) & 0xFFFFu)          // tensor_dim1[31:16]
          | ((tile_d0 & 0xFFFFu) << 16));              // tile_dim0
    g1[4] = (int)(tile_d1 & 0xFFFFu);                  // tile_dim1 (tile_dim2=0)
    g1[5] = (int)stride_elems;                         // tensor_dim0_stride lo32
    g1[6] = 0;                                         // stride hi / dim1_stride
    g1[7] = 0;
    i32x4 z4 = {};
    i32x8 z8 = {};
    __builtin_amdgcn_tensor_load_to_lds(g0, g1, z4, z4, z8, 0);
}
#endif

// ---------------------------------------------------------------------------
// Kernel 1: X = HS @ W + b, fused RoPE, emit f16 q/k into workspace.
// Block = 128 threads (4 wave32). Block tile: 64 rows (M) x 64 cols (N).
// Tiles are DMA'd into LDS as f32 by the Tensor Data Mover (wave 0 issues,
// s_wait_tensorcnt, block barrier); fragments convert f32->f16 on ds read.
// ---------------------------------------------------------------------------
__global__ __launch_bounds__(128) void proj_rope_kernel(
    const float* __restrict__ A,      // [4096][1024] last_hidden_state
    const float* __restrict__ W,      // [1024][2048]
    const float* __restrict__ bias,   // [2048]
    _Float16* __restrict__ qws,       // [B*H][L][D] f16
    _Float16* __restrict__ kws)       // [B*H][L][D] f16
{
    __shared__ float Ats[64][32];     // A tile  [m_local][k_local] f32 (8 KB)
    __shared__ float Wts[32][64];     // W tile  [k_local][n_local] f32 (8 KB)
    __shared__ float Xs[4][16][64];   // epilogue staging (16 KB)

    const int t    = threadIdx.x;
    const int lane = t & 31;
    const int wv   = t >> 5;
    const int m0   = blockIdx.x * 64;
    const int n0   = blockIdx.y * 64;

#if USE_TDM
    const unsigned ldsA = (unsigned)(uintptr_t)&Ats[0][0];
    const unsigned ldsW = (unsigned)(uintptr_t)&Wts[0][0];
#endif

    v8f acc[4] = {};

    for (int k0 = 0; k0 < HID; k0 += 32) {
#if USE_TDM
        if (wv == 0) {
            // A tile: 32 wide x 64 rows from [4096][1024]
            tdm_load_tile_f32(ldsA, A + (size_t)m0 * HID + k0,
                              32u, 64u, (unsigned)HID, (unsigned)(BB * LL),
                              (unsigned)HID);
            // W tile: 64 wide x 32 rows from [1024][2048]
            tdm_load_tile_f32(ldsW, W + (size_t)k0 * NCOL + n0,
                              64u, 32u, (unsigned)NCOL, (unsigned)HID,
                              (unsigned)NCOL);
            __builtin_amdgcn_s_wait_tensorcnt(0);
        }
#else
        {   // fallback: cooperative float4 copies (no transpose needed)
            const int m  = t >> 1;
            const int kh = (t & 1) * 16;
            const float4* srcA = (const float4*)(A + (size_t)(m0 + m) * HID + k0 + kh);
            float4* dstA = (float4*)&Ats[m][kh];
#pragma unroll
            for (int j = 0; j < 4; ++j) dstA[j] = srcA[j];
            const int k  = t >> 2;
            const int nh = (t & 3) * 16;
            const float4* srcW = (const float4*)(W + (size_t)(k0 + k) * NCOL + n0 + nh);
            float4* dstW = (float4*)&Wts[k][nh];
#pragma unroll
            for (int j = 0; j < 4; ++j) dstW[j] = srcW[j];
        }
#endif
        if (k0 + 32 < HID) {   // CDNA5 global_prefetch_b8 on next tiles
            __builtin_prefetch(A + (size_t)(m0 + (t >> 1)) * HID + k0 + 32 + (t & 1) * 16, 0, 3);
            __builtin_prefetch(W + (size_t)(k0 + 32 + (t >> 2)) * NCOL + n0 + (t & 3) * 16, 0, 3);
        }
        __syncthreads();

        // --- A fragment: contiguous K per lane-half (same order as B) --------------------
        // lane<16: M=lane,    K = 0..15
        // lane>=16: M=lane-16, K = 16..31
        const int row = wv * 16 + (lane & 15);
        const int kb  = (lane >> 4) * 16;
        Frag16 a;
        {
            const float4 f0 = *(const float4*)&Ats[row][kb];
            const float4 f1 = *(const float4*)&Ats[row][kb + 4];
            const float4 f2 = *(const float4*)&Ats[row][kb + 8];
            const float4 f3 = *(const float4*)&Ats[row][kb + 12];
            a.e[0]  = (_Float16)f0.x; a.e[1]  = (_Float16)f0.y;
            a.e[2]  = (_Float16)f0.z; a.e[3]  = (_Float16)f0.w;
            a.e[4]  = (_Float16)f1.x; a.e[5]  = (_Float16)f1.y;
            a.e[6]  = (_Float16)f1.z; a.e[7]  = (_Float16)f1.w;
            a.e[8]  = (_Float16)f2.x; a.e[9]  = (_Float16)f2.y;
            a.e[10] = (_Float16)f2.z; a.e[11] = (_Float16)f2.w;
            a.e[12] = (_Float16)f3.x; a.e[13] = (_Float16)f3.y;
            a.e[14] = (_Float16)f3.z; a.e[15] = (_Float16)f3.w;
        }

        // --- B fragments: lane = N col, 16 consecutive K per lane-half -
        const int kb2 = (lane >> 4) * 16;
#pragma unroll
        for (int c = 0; c < 4; ++c) {
            const int n = c * 16 + (lane & 15);
            Frag16 b;
#pragma unroll
            for (int e = 0; e < 16; ++e)
                b.e[e] = (_Float16)Wts[kb2 + e][n];
            acc[c] = __builtin_amdgcn_wmma_f32_16x16x32_f16(
                false, a.v, false, b.v, (short)0, acc[c], false, false);
        }
        __syncthreads();
    }

    // --- epilogue: bias, stage to LDS for cross-lane RoPE pairing ------
#pragma unroll
    for (int c = 0; c < 4; ++c) {
        const int col = c * 16 + (lane & 15);
        const float bv = bias[n0 + col];
#pragma unroll
        for (int g = 0; g < 8; ++g) {
            const int r = g + 8 * (lane >> 4);      // C-matrix row layout
            Xs[wv][r][col] = acc[c][g] + bv;
        }
    }
    __syncthreads();

    // --- RoPE: out[r] = t1*cos - t2*sin (r<32), t2*cos + t1*sin (r>=32)
    const int db  = n0 >> 6;          // d-block index 0..31 within head space
    const int hh  = db >> 1;
    const int isK = db & 1;
    _Float16* dst_base = isK ? kws : qws;
    const float theta = __powf(10000.f, -(float)lane * (1.f / 32.f));
    for (int m = 0; m < 16; ++m) {
        const int rowg = m0 + wv * 16 + m;
        const int bidx = rowg >> 9;               // / L
        const int l    = rowg & (LL - 1);         // % L
        float s, c;
        __sincosf((float)l * theta, &s, &c);
        const float t1 = Xs[wv][m][2 * lane];
        const float t2 = Xs[wv][m][2 * lane + 1];
        _Float16* dst = dst_base + ((size_t)(bidx * HH + hh) * LL + l) * DD;
        dst[lane]      = (_Float16)(t1 * c - t2 * s);
        dst[lane + 32] = (_Float16)(t2 * c + t1 * s);
    }
}

// ---------------------------------------------------------------------------
// Kernel 2: logits[bh][m][n] = (q[m] . k[n]) with pad + causal mask, *0.125.
// One wave per block, 64x64 output tile = 4x4 WMMA accumulators, K=64.
// Fragments load as contiguous 16B chunks directly from the f16 workspace.
// ---------------------------------------------------------------------------
__global__ __launch_bounds__(32) void attn_logits_kernel(
    const _Float16* __restrict__ qws,   // [B*H][L][D]
    const _Float16* __restrict__ kws,   // [B*H][L][D]
    const float* __restrict__ amask,    // [B][L]
    float* __restrict__ out)            // [B*H][L][L]
{
    const int lane = threadIdx.x & 31;
    const int nt0  = blockIdx.x * 64;
    const int mt0  = blockIdx.y * 64;
    const int bh   = blockIdx.z;            // 0..127
    const int bidx = bh >> 4;

    const _Float16* qb = qws + (size_t)bh * LL * DD;
    const _Float16* kb = kws + (size_t)bh * LL * DD;

    v8f acc[4][4] = {};

#pragma unroll
    for (int kblk = 0; kblk < 2; ++kblk) {
        const int ka  = kblk * 32 + (lane >> 4) * 8;    // A-frag K base
        const int kn  = kblk * 32 + (lane >> 4) * 8;    // B-frag K base
        Frag16 af[4], bf[4];
#pragma unroll
        for (int i = 0; i < 4; ++i) {
            const int mrow = mt0 + i * 16 + (lane & 15);
            af[i].h[0] = *(const v8h*)(qb + (size_t)mrow * DD + ka);
            af[i].h[1] = *(const v8h*)(qb + (size_t)mrow * DD + ka + 16);
            const int ncol = nt0 + i * 16 + (lane & 15);
            bf[i].h[0] = *(const v8h*)(kb + (size_t)ncol * DD + kn);
            bf[i].h[1] = *(const v8h*)(kb + (size_t)ncol * DD + kn + 16);
        }
#pragma unroll
        for (int i = 0; i < 4; ++i)
#pragma unroll
            for (int j = 0; j < 4; ++j)
                acc[i][j] = __builtin_amdgcn_wmma_f32_16x16x32_f16(
                    false, af[i].v, false, bf[j].v, (short)0, acc[i][j], false, false);
    }

    // --- epilogue: pad mask, strict-lower causal, scale by 1/sqrt(64) --
#pragma unroll
    for (int j = 0; j < 4; ++j) {
        const int col = nt0 + j * 16 + (lane & 15);
        const float pad = amask[bidx * LL + col];
        const float padoff = (1.f - pad) * NEGV;
#pragma unroll
        for (int i = 0; i < 4; ++i) {
#pragma unroll
            for (int g = 0; g < 8; ++g) {
                const int row = mt0 + i * 16 + g + 8 * (lane >> 4);
                float v = acc[i][j][g];
                v = v * pad - padoff;
                if (col < row) v -= NEGV;       // tril(-1) mask
                out[((size_t)bh * LL + row) * LL + col] = v * 0.125f;
            }
        }
    }
}

// ---------------------------------------------------------------------------
extern "C" void kernel_launch(void* const* d_in, const int* in_sizes, int n_in,
                              void* d_out, int out_size, void* d_ws, size_t ws_size,
                              hipStream_t stream) {
    (void)in_sizes; (void)n_in; (void)out_size; (void)ws_size;
    const float* hs   = (const float*)d_in[0];   // [8][512][1024]
    const float* mask = (const float*)d_in[1];   // [8][512]
    const float* W    = (const float*)d_in[2];   // [1024][2048]
    const float* bias = (const float*)d_in[3];   // [2048]
    float* out = (float*)d_out;                  // [8][16][512][512]

    _Float16* qws = (_Float16*)d_ws;                          // 8 MB
    _Float16* kws = qws + (size_t)BB * HH * LL * DD;          // +8 MB (ws >= 16 MB)

    dim3 grid1((BB * LL) / 64, NCOL / 64);        // 64 x 32
    proj_rope_kernel<<<grid1, 128, 0, stream>>>(hs, W, bias, qws, kws);

    dim3 grid2(LL / 64, LL / 64, BB * HH);        // 8 x 8 x 128
    attn_logits_kernel<<<grid2, 32, 0, stream>>>(qws, kws, mask, out);
}
